// MahalanobisOODDetector_67843303408176
// MI455X (gfx1250) — compile-verified, matrix operand order checked by reading, not run on hardware
//
#include <hip/hip_runtime.h>
#include <math.h>

typedef __attribute__((ext_vector_type(16))) __bf16 v16bf;
typedef __attribute__((ext_vector_type(8)))  __bf16 v8bf;
typedef __attribute__((ext_vector_type(8)))  float  v8f;

#define DIM   512
#define NCLS  64
#define NROW  131072
#define LDX   520              // padded LDS row stride (bf16 elems): 1040B, shifts 4 banks/row

union V16U { v16bf v; v8bf h[2]; };

// ---------------- prep kernel 1: P (f32) -> Pbf (bf16) ----------------
__global__ void cvt_p_kernel(const float* __restrict__ P, __bf16* __restrict__ Pbf) {
    int i = (blockIdx.x * blockDim.x + threadIdx.x) * 4;
    float4 f = *(const float4*)(P + i);
    Pbf[i + 0] = (__bf16)f.x;
    Pbf[i + 1] = (__bf16)f.y;
    Pbf[i + 2] = (__bf16)f.z;
    Pbf[i + 3] = (__bf16)f.w;
}

// ------------- prep kernel 2: muP = mu @ P (f32), muPbf, muPmu -------------
__global__ void mup_kernel(const float* __restrict__ mu, const float* __restrict__ P,
                           __bf16* __restrict__ muPbf, float* __restrict__ muPmu) {
    __shared__ float mus[DIM];
    __shared__ float red[256];
    const int c = blockIdx.x, t = threadIdx.x;
    mus[t]       = mu[c * DIM + t];
    mus[t + 256] = mu[c * DIM + t + 256];
    __syncthreads();
    float a0 = 0.f, a1 = 0.f;
    for (int k = 0; k < DIM; ++k) {
        float m = mus[k];
        a0 += m * P[k * DIM + t];         // coalesced over t
        a1 += m * P[k * DIM + t + 256];
    }
    muPbf[c * DIM + t]       = (__bf16)a0;
    muPbf[c * DIM + t + 256] = (__bf16)a1;
    red[t] = a0 * mus[t] + a1 * mus[t + 256];
    __syncthreads();
    for (int s = 128; s > 0; s >>= 1) {
        if (t < s) red[t] += red[t + s];
        __syncthreads();
    }
    if (t == 0) muPmu[c] = red[0];
}

// ---------------- main kernel: fused Mahalanobis + min ----------------
__global__ __launch_bounds__(256) void maha_kernel(
    const float*  __restrict__ X,
    const __bf16* __restrict__ Pbf,     // [512][512] bf16, symmetric
    const __bf16* __restrict__ muPbf,   // [64][512] bf16
    const float*  __restrict__ muPmu,   // [64]
    float*        __restrict__ out) {

    __shared__ __bf16 Xs[128 * LDX];    // 130 KB: block's 128x512 X tile (bf16)

    const int t    = threadIdx.x;
    const int base = blockIdx.x * 128;

    // ---- stage X tile f32 -> bf16 LDS (coalesced float4 loads) ----
    for (int it = 0; it < 64; ++it) {
        int flat = (it * 256 + t) * 4;
        int row  = flat >> 9;           // /512
        int col  = flat & 511;
        float4 f = *(const float4*)(X + (size_t)(base + row) * DIM + col);
        __bf16* dst = &Xs[row * LDX + col];
        dst[0] = (__bf16)f.x; dst[1] = (__bf16)f.y;
        dst[2] = (__bf16)f.z; dst[3] = (__bf16)f.w;
    }
    __syncthreads();

    const int wave = t >> 5;            // 0..7, owns rows wave*16..+15
    const int lane = t & 31;
    const int ln16 = lane & 15;
    const int hi   = lane >> 4;         // half-wave select
    const int rowA = wave * 16 + ln16;  // A-fragment row for this lane
    const int koff = hi * 8;            // K-chunk offset per 16-bit A/B layout

    float xpx[8];
#pragma unroll
    for (int r = 0; r < 8; ++r) xpx[r] = 0.f;

    // ================= Y = X @ P, fused x·Y for xPx =================
    for (int p = 0; p < 4; ++p) {       // 4 passes x 128 columns
        v8f acc[8];
#pragma unroll
        for (int ct = 0; ct < 8; ++ct) acc[ct] = (v8f){0,0,0,0,0,0,0,0};
        const int colBase = p * 128;

        for (int k0 = 0; k0 < DIM; k0 += 32) {
            V16U A;
            A.h[0] = *(const v8bf*)&Xs[rowA * LDX + k0 + koff];
            A.h[1] = *(const v8bf*)&Xs[rowA * LDX + k0 + 16 + koff];
#pragma unroll
            for (int ct = 0; ct < 8; ++ct) {
                // symmetric P: B column (colBase+ct*16+ln16) == contiguous row of P
                const __bf16* bp = Pbf + (size_t)(colBase + ct * 16 + ln16) * DIM + k0 + koff;
                V16U B;
                B.h[0] = *(const v8bf*)bp;
                B.h[1] = *(const v8bf*)(bp + 16);
                acc[ct] = __builtin_amdgcn_wmma_f32_16x16x32_bf16(
                    false, A.v, false, B.v, (short)0, acc[ct], false, false);
            }
        }
        // epilogue: xPx partials, directly in the C-matrix layout
#pragma unroll
        for (int ct = 0; ct < 8; ++ct) {
            int col = colBase + ct * 16 + ln16;
#pragma unroll
            for (int r = 0; r < 8; ++r) {
                int m = wave * 16 + r + 8 * hi;
                xpx[r] += acc[ct][r] * (float)Xs[m * LDX + col];
            }
        }
    }
    // butterfly-sum xPx over the 16 columns held across each lane half
#pragma unroll
    for (int r = 0; r < 8; ++r) {
        float v = xpx[r];
        v += __shfl_xor(v, 1, 32);
        v += __shfl_xor(v, 2, 32);
        v += __shfl_xor(v, 4, 32);
        v += __shfl_xor(v, 8, 32);
        xpx[r] = v;                      // all lanes of each half hold row sum
    }

    // ================= S = X @ muP^T (64 classes) =================
    v8f sacc[4];
#pragma unroll
    for (int ct = 0; ct < 4; ++ct) sacc[ct] = (v8f){0,0,0,0,0,0,0,0};

    for (int k0 = 0; k0 < DIM; k0 += 32) {
        V16U A;
        A.h[0] = *(const v8bf*)&Xs[rowA * LDX + k0 + koff];
        A.h[1] = *(const v8bf*)&Xs[rowA * LDX + k0 + 16 + koff];
#pragma unroll
        for (int ct = 0; ct < 4; ++ct) {
            const __bf16* bp = muPbf + (size_t)(ct * 16 + ln16) * DIM + k0 + koff;
            V16U B;
            B.h[0] = *(const v8bf*)bp;
            B.h[1] = *(const v8bf*)(bp + 16);
            sacc[ct] = __builtin_amdgcn_wmma_f32_16x16x32_bf16(
                false, A.v, false, B.v, (short)0, sacc[ct], false, false);
        }
    }

    // ============ distances + min over classes ============
    float mupmu_l[4];
#pragma unroll
    for (int ct = 0; ct < 4; ++ct) mupmu_l[ct] = muPmu[ct * 16 + ln16];

    float dmin[8];
#pragma unroll
    for (int r = 0; r < 8; ++r) dmin[r] = 3.0e38f;

#pragma unroll
    for (int ct = 0; ct < 4; ++ct) {
#pragma unroll
        for (int r = 0; r < 8; ++r) {
            float d2 = xpx[r] - 2.0f * sacc[ct][r] + mupmu_l[ct];
            float d  = sqrtf(fmaxf(d2, 0.0f));
            dmin[r]  = fminf(dmin[r], d);
        }
    }
    // min over the 16 classes per lane half, then write
#pragma unroll
    for (int r = 0; r < 8; ++r) {
        float v = dmin[r];
        v = fminf(v, __shfl_xor(v, 1, 32));
        v = fminf(v, __shfl_xor(v, 2, 32));
        v = fminf(v, __shfl_xor(v, 4, 32));
        v = fminf(v, __shfl_xor(v, 8, 32));
        if (ln16 == 0)
            out[base + wave * 16 + r + 8 * hi] = v;
    }
}

// ---------------- launcher ----------------
extern "C" void kernel_launch(void* const* d_in, const int* in_sizes, int n_in,
                              void* d_out, int out_size, void* d_ws, size_t ws_size,
                              hipStream_t stream) {
    const float* features = (const float*)d_in[0];   // [N, 512]
    const float* means    = (const float*)d_in[1];   // [64, 512]
    const float* prec     = (const float*)d_in[2];   // [512, 512] symmetric

    // workspace layout
    __bf16* Pbf   = (__bf16*)d_ws;                                  // 512 KB
    __bf16* muPbf = (__bf16*)((char*)d_ws + DIM * DIM * 2);         //  64 KB
    float*  muPmu = (float*)((char*)d_ws + DIM * DIM * 2 + NCLS * DIM * 2); // 256 B

    cvt_p_kernel<<<(DIM * DIM) / (256 * 4), 256, 0, stream>>>(prec, Pbf);
    mup_kernel<<<NCLS, 256, 0, stream>>>(means, prec, muPbf, muPmu);
    maha_kernel<<<NROW / 128, 256, 0, stream>>>(features, Pbf, muPbf, muPmu,
                                                (float*)d_out);
}
